// Encoder_sf_27496380629559
// MI455X (gfx1250) — matrix operand order repaired and checked by reading, not compiled, not run on hardware
//
#include <hip/hip_runtime.h>
#include <math.h>

// ---------------------------------------------------------------------------
// Types for CDNA5 WMMA (wave32): v_wmma_f32_16x16x32_bf16
// ---------------------------------------------------------------------------
typedef __bf16 bf16_t;
typedef __attribute__((ext_vector_type(16))) __bf16 v16bf;
typedef __attribute__((ext_vector_type(8)))  __bf16 v8bf;
typedef __attribute__((ext_vector_type(8)))  float  v8f;

static __device__ __forceinline__ unsigned short f2bf(float f) {
  unsigned int u = __float_as_uint(f);                 // round-to-nearest-even
  return (unsigned short)((u + 0x7FFFu + ((u >> 16) & 1u)) >> 16);
}
static __device__ __forceinline__ float bf2f(unsigned short h) {
  return __uint_as_float(((unsigned int)h) << 16);
}
static __device__ __forceinline__ bf16_t f2bf16t(float f) {
  union { unsigned short u; bf16_t h; } cv;
  cv.u = f2bf(f);
  return cv.h;
}
static __device__ __forceinline__ float elu1(float z) {
  return z > 0.f ? z : (expf(z) - 1.f);
}

// ---------------------------------------------------------------------------
// Elementwise converters
// ---------------------------------------------------------------------------
__global__ void cvt_f32_bf16(const float* __restrict__ in,
                             unsigned short* __restrict__ out, int n) {
  int i = blockIdx.x * blockDim.x + threadIdx.x;
  if (i < n) out[i] = f2bf(in[i]);
}

// W (Kreal x N, f32, row-major) -> Wt (N x Kpad, bf16), zero-padded past Kreal.
__global__ void cvt_w_pad_t(const float* __restrict__ W,
                            unsigned short* __restrict__ wt,
                            int Kreal, int Kpad, int N) {
  int i = blockIdx.x * blockDim.x + threadIdx.x;
  if (i >= N * Kpad) return;
  int k = i % Kpad;
  int n = i / Kpad;
  wt[i] = (k < Kreal) ? f2bf(W[(size_t)k * N + n]) : (unsigned short)0;
}

__global__ void zero_f32(float* __restrict__ p, int n) {
  int i = blockIdx.x * blockDim.x + threadIdx.x;
  if (i < n) p[i] = 0.f;
}

// ---------------------------------------------------------------------------
// Spiral conv + bias + ELU, one kernel for all 4 levels.
//   M = 32*V rows, K = 12*C (padded to Kpad, mult of 32), N = Cout.
//   grid.x = M/16 (one 16-row A tile per block, gathered once into LDS)
//   block  = 32 * (N/16) threads: each wave owns one 16x16 output tile.
// ---------------------------------------------------------------------------
__global__ void conv_spiral_wmma(const unsigned short* __restrict__ xin,  // (32,V,C) bf16
                                 const int*           __restrict__ spiral,// (V,12)
                                 const unsigned short* __restrict__ wt,   // (N,Kpad) bf16
                                 const float*         __restrict__ bias,  // (N)
                                 unsigned short*      __restrict__ out,   // (32,V,N) bf16
                                 int V, int C, int Kreal, int Kpad, int N)
{
  extern __shared__ unsigned short lds[];               // [16][Kpad] bf16 bits
  const int r0     = blockIdx.x * 16;
  const int vmask  = V - 1;
  const int vshift = __ffs(V) - 1;                      // V is a power of two
  const bool cpow2 = (C & (C - 1)) == 0;
  const int cmask  = C - 1;
  const int cshift = __ffs(C) - 1;

  // ---- Stage gathered A tile: lds[m][k] = x[b, spiral[v][k/C], k%C] ----
  for (int m = 0; m < 16; ++m) {
    int r = r0 + m;
    int b = r >> vshift;
    int v = r & vmask;
    const int*            srow = spiral + v * 12;
    const unsigned short* xrow = xin + (size_t)b * V * C;
    for (int k = threadIdx.x; k < Kpad; k += blockDim.x) {
      unsigned short val = 0;
      if (k < Kreal) {
        int s, c;
        if (cpow2) { s = k >> cshift;       c = k & cmask;  }
        else       { s = (k * 21846) >> 16; c = k - s * 3;  }  // k/3 for C==3
        int idx = srow[s];
        val = xrow[idx * C + c];
      }
      lds[m * Kpad + k] = val;
    }
  }
  __syncthreads();

  // ---- WMMA mainloop: one 16x16 tile per wave ----
  const int lane  = threadIdx.x & 31;
  const int wave  = threadIdx.x >> 5;
  const int nn    = lane & 15;        // row of A fragment / col of B,C,D
  const int khalf = lane >> 4;
  const int col   = wave * 16 + nn;

  const bf16_t* arow = (const bf16_t*)lds + nn * Kpad;          // A row (16B-aligned)
  const bf16_t* wcol = (const bf16_t*)wt  + (size_t)col * Kpad; // B col, contiguous K

  v8f acc = {};
  for (int k0 = 0; k0 < Kpad; k0 += 32) {
    v16bf a, bm;
    // A 16-bit layout: e0-7 -> k = k0 + 8*khalf + e ; e8-15 -> k = k0+16+8*khalf+(e-8)
    ((v8bf*)&a)[0]  = *(const v8bf*)(arow + k0 + khalf * 8);
    ((v8bf*)&a)[1]  = *(const v8bf*)(arow + k0 + 16 + khalf * 8);
    // B 16-bit layout: e0-7 -> k = k0 + 8*khalf + e ; e8-15 -> k = k0+16+8*khalf+(e-8)
    ((v8bf*)&bm)[0] = *(const v8bf*)(wcol + k0 + khalf * 8);
    ((v8bf*)&bm)[1] = *(const v8bf*)(wcol + k0 + 16 + khalf * 8);
    acc = __builtin_amdgcn_wmma_f32_16x16x32_bf16(false, a, false, bm,
                                                  (short)0, acc, false, false);
  }

  // ---- Epilogue: bias + ELU -> bf16.  C/D: row = j + 8*khalf, col = lane%16 ----
  float bv = bias[col];
#pragma unroll
  for (int j = 0; j < 8; ++j) {
    int r = r0 + j + 8 * khalf;
    out[(size_t)r * N + col] = f2bf(elu1(acc[j] + bv));
  }
}

// ---------------------------------------------------------------------------
// Mesh pool: out[b,vo,c] = sum_k dw[vo,k] * in[b, didx[vo,k], c]   (bf16 io)
// ---------------------------------------------------------------------------
__global__ void mesh_pool_k(const unsigned short* __restrict__ in,
                            const int*            __restrict__ didx,
                            const float*          __restrict__ dw,
                            unsigned short*       __restrict__ out,
                            int Vin, int Vout, int Cc, int total)
{
  int tid = blockIdx.x * blockDim.x + threadIdx.x;
  if (tid >= total) return;
  int cshift  = __ffs(Cc) - 1;
  int voshift = __ffs(Vout) - 1;
  int c  = tid & (Cc - 1);
  int t2 = tid >> cshift;
  int vo = t2 & (Vout - 1);
  int b  = t2 >> voshift;
  const int*   ip = didx + vo * 3;
  const float* wp = dw   + vo * 3;
  float s = 0.f;
#pragma unroll
  for (int k = 0; k < 3; ++k) {
    int idx = ip[k];
    s += wp[k] * bf2f(in[((size_t)b * Vin + idx) * Cc + c]);
  }
  out[tid] = f2bf(s);
}

// ---------------------------------------------------------------------------
// FC1: (32 x 32768) @ (32768 x 512), K-split 8 with f32 atomic reduction.
// A rows are contiguous bf16 -> direct global 16B fragment loads.
// B converted f32->bf16 on the fly (loads are lane-coalesced over n).
// ---------------------------------------------------------------------------
__global__ void fc1_wmma(const unsigned short* __restrict__ X,    // (32,32768) bf16
                         const float*         __restrict__ Wl1,   // (32768,512) f32
                         float*               __restrict__ acc_out) // (32,512) f32
{
  const int KSPLIT = 8, KCH = 32768 / KSPLIT;
  int bid = blockIdx.x;
  int ks  = bid % KSPLIT;
  int t   = bid / KSPLIT;
  int nt  = t % 32;
  int mt  = t / 32;
  int lane  = threadIdx.x & 31;
  int nn    = lane & 15;
  int khalf = lane >> 4;
  int col   = nt * 16 + nn;
  int r0    = mt * 16;
  const bf16_t* arow = (const bf16_t*)X + (size_t)(r0 + nn) * 32768;
  int kbase = ks * KCH;

  v8f acc = {};
  for (int kc = 0; kc < KCH; kc += 32) {
    int k0 = kbase + kc;
    v16bf a, bm;
    ((v8bf*)&a)[0] = *(const v8bf*)(arow + k0 + khalf * 8);
    ((v8bf*)&a)[1] = *(const v8bf*)(arow + k0 + 16 + khalf * 8);
#pragma unroll
    for (int e = 0; e < 16; ++e)
      bm[e] = f2bf16t(Wl1[(size_t)(k0 + khalf * 8 + e + (e >= 8 ? 8 : 0)) * 512 + col]);
    acc = __builtin_amdgcn_wmma_f32_16x16x32_bf16(false, a, false, bm,
                                                  (short)0, acc, false, false);
  }
#pragma unroll
  for (int j = 0; j < 8; ++j)
    atomicAdd(&acc_out[(size_t)(r0 + j + 8 * khalf) * 512 + col], acc[j]);
}

// bias + ELU; emit f32 tuple output [0] and bf16 copy for FC2.
__global__ void fc1_fin(const float* __restrict__ acc,
                        const float* __restrict__ bl1,
                        float*          __restrict__ out_x,
                        unsigned short* __restrict__ x1b)
{
  int i = blockIdx.x * blockDim.x + threadIdx.x;
  if (i >= 32 * 512) return;
  float v = elu1(acc[i] + bl1[i & 511]);
  out_x[i] = v;
  x1b[i]   = f2bf(v);
}

// FC2: (32 x 512) @ (512 x 64) + bl2 -> f32 tuple output [1]
__global__ void fc2_wmma(const unsigned short* __restrict__ X1,  // (32,512) bf16
                         const float* __restrict__ Wl2,          // (512,64) f32
                         const float* __restrict__ bl2,
                         float*       __restrict__ Y)            // (32,64) f32
{
  int bid = blockIdx.x;           // 8 blocks: 2 M-tiles x 4 N-tiles
  int nt  = bid & 3;
  int mt  = bid >> 2;
  int lane  = threadIdx.x & 31;
  int nn    = lane & 15;
  int khalf = lane >> 4;
  int col   = nt * 16 + nn;
  int r0    = mt * 16;
  const bf16_t* arow = (const bf16_t*)X1 + (size_t)(r0 + nn) * 512;

  v8f acc = {};
  for (int k0 = 0; k0 < 512; k0 += 32) {
    v16bf a, bm;
    ((v8bf*)&a)[0] = *(const v8bf*)(arow + k0 + khalf * 8);
    ((v8bf*)&a)[1] = *(const v8bf*)(arow + k0 + 16 + khalf * 8);
#pragma unroll
    for (int e = 0; e < 16; ++e)
      bm[e] = f2bf16t(Wl2[(size_t)(k0 + khalf * 8 + e + (e >= 8 ? 8 : 0)) * 64 + col]);
    acc = __builtin_amdgcn_wmma_f32_16x16x32_bf16(false, a, false, bm,
                                                  (short)0, acc, false, false);
  }
  float bv = bl2[col];
#pragma unroll
  for (int j = 0; j < 8; ++j)
    Y[(size_t)(r0 + j + 8 * khalf) * 64 + col] = acc[j] + bv;
}

// ---------------------------------------------------------------------------
// Host-side orchestration
// ---------------------------------------------------------------------------
extern "C" void kernel_launch(void* const* d_in, const int* in_sizes, int n_in,
                              void* d_out, int out_size, void* d_ws, size_t ws_size,
                              hipStream_t stream)
{
  (void)in_sizes; (void)n_in; (void)out_size; (void)ws_size;

  static const int VL[5]    = {32768, 8192, 2048, 512, 128};
  static const int CH[5]    = {3, 32, 64, 128, 256};
  static const int Kreal[4] = {36, 384, 768, 1536};
  static const int Kpad[4]  = {64, 384, 768, 1536};

  // Inputs in setup_inputs() dict order:
  const float* x = (const float*)d_in[0];
  const int*   spiral[4]; const int* didx[4];
  const float* dw[4]; const float* W[4]; const float* bconv[4];
  for (int i = 0; i < 4; ++i) {
    spiral[i] = (const int*)  d_in[1 + 5 * i];
    didx[i]   = (const int*)  d_in[2 + 5 * i];
    dw[i]     = (const float*)d_in[3 + 5 * i];
    W[i]      = (const float*)d_in[4 + 5 * i];
    bconv[i]  = (const float*)d_in[5 + 5 * i];
  }
  const float* Wl1 = (const float*)d_in[21];
  const float* bl1 = (const float*)d_in[22];
  const float* Wl2 = (const float*)d_in[23];
  const float* bl2 = (const float*)d_in[24];

  // Workspace layout (bytes)
  char* ws = (char*)d_ws;
  size_t off = 0;
  unsigned short* xb = (unsigned short*)(ws + off); off += (size_t)32 * 32768 * 3 * 2; // 6.3MB
  unsigned short* wtp[4];
  for (int i = 0; i < 4; ++i) {
    wtp[i] = (unsigned short*)(ws + off);
    off += (size_t)CH[i + 1] * Kpad[i] * 2;
  }
  unsigned short* bufA   = (unsigned short*)(ws + off); off += 67108864; // conv outputs
  unsigned short* bufB   = (unsigned short*)(ws + off); off += 16777216; // pool outputs
  float*          fc1acc = (float*)(ws + off);          off += 65536;
  unsigned short* x1b    = (unsigned short*)(ws + off); off += 32768;

  // 1) x -> bf16
  {
    int n = 32 * 32768 * 3;
    cvt_f32_bf16<<<(n + 255) / 256, 256, 0, stream>>>(x, xb, n);
  }
  // 2) conv weights -> transposed, padded bf16
  for (int i = 0; i < 4; ++i) {
    int n = CH[i + 1] * Kpad[i];
    cvt_w_pad_t<<<(n + 255) / 256, 256, 0, stream>>>(W[i], wtp[i],
                                                     Kreal[i], Kpad[i], CH[i + 1]);
  }
  // 3) four levels of spiral-conv (+ELU) and mesh-pool
  for (int i = 0; i < 4; ++i) {
    const unsigned short* in = (i == 0) ? xb : bufB;
    int V = VL[i], C = CH[i], N = CH[i + 1];
    dim3 grid(2 * V);                       // (32*V)/16 row tiles
    dim3 block(32 * (N / 16));              // one wave per 16-col tile
    size_t smem = (size_t)16 * Kpad[i] * 2; // staged A tile
    conv_spiral_wmma<<<grid, block, smem, stream>>>(in, spiral[i], wtp[i], bconv[i],
                                                    bufA, V, C, Kreal[i], Kpad[i], N);
    int total = 32 * VL[i + 1] * N;
    mesh_pool_k<<<(total + 255) / 256, 256, 0, stream>>>(bufA, didx[i], dw[i], bufB,
                                                         V, VL[i + 1], N, total);
  }
  // 4) FC1 with K-split atomic reduction
  zero_f32<<<(16384 + 255) / 256, 256, 0, stream>>>(fc1acc, 16384);
  fc1_wmma<<<2 * 32 * 8, 32, 0, stream>>>(bufB, Wl1, fc1acc);
  fc1_fin<<<(16384 + 255) / 256, 256, 0, stream>>>(fc1acc, bl1, (float*)d_out, x1b);
  // 5) FC2 -> tuple output [1]
  fc2_wmma<<<8, 32, 0, stream>>>(x1b, Wl2, bl2, (float*)d_out + 32 * 512);
}
